// IAIDNN_45878840655924
// MI455X (gfx1250) — matrix operand order repaired and checked by reading, not compiled, not run on hardware
//
#include <hip/hip_runtime.h>
#include <math.h>

typedef __attribute__((ext_vector_type(16))) __bf16 v16bf;
typedef __attribute__((ext_vector_type(8)))  float  v8f;

#define NBATCH 512
#define NOISE_POW 1.0f
#define BSS_POW 16.0f
#define EPSV 1e-8f
#define LDB 68   // bf16 operand leading dim: 136B rows -> 8B-aligned, conflict-free (34-bank step)

struct cpx { float r, i; };
__device__ __forceinline__ cpx cadd(cpx a, cpx b){ return {a.r+b.r, a.i+b.i}; }
__device__ __forceinline__ cpx cmul(cpx a, cpx b){ return {a.r*b.r - a.i*b.i, a.r*b.i + a.i*b.r}; }
// conj(a) * b
__device__ __forceinline__ cpx cmulc(cpx a, cpx b){ return {a.r*b.r + a.i*b.i, a.r*b.i - a.i*b.r}; }
// 1 / (a + EPS)  (EPS added to real part, as jnp does for complex d + real scalar)
__device__ __forceinline__ cpx crecip_eps(cpx d){
  float dr = d.r + EPSV;
  float s = 1.0f / (dr*dr + d.i*d.i);
  return { dr*s, -d.i*s };
}

// ---- WMMA bf16 16x16x32 fragment helpers (wave32 layouts per CDNA5 ISA 7.12.2) ----
// A: 16x32 (MxK) row-major, ld=LDB. lanes 0-15: row=lane, K={k0..k0+7, k0+16..k0+23};
// lanes 16-31: row=lane-16, K={k0+8..k0+15, k0+24..k0+31}. Two contiguous 8-elem runs -> b64 loads.
__device__ __forceinline__ v16bf frag_a(const __bf16* A, int m0, int k0, int lane){
  const __bf16* p = A + (m0 + (lane & 15)) * LDB + k0 + ((lane & 16) ? 8 : 0);
  union { unsigned long long u[4]; v16bf v; } U;
  U.u[0] = *(const unsigned long long*)(p);
  U.u[1] = *(const unsigned long long*)(p + 4);
  U.u[2] = *(const unsigned long long*)(p + 16);
  U.u[3] = *(const unsigned long long*)(p + 20);
  return U.v;
}
// B stored TRANSPOSED (Bt[n][k], ld=LDB). lanes 0-15: col n=lane, K={k0..k0+7, k0+16..k0+23};
// lanes 16-31: col n=lane-16, K={k0+8..k0+15, k0+24..k0+31}. Two contiguous 8-elem runs -> b64 loads.
__device__ __forceinline__ v16bf frag_bt(const __bf16* Bt, int n0, int k0, int lane){
  const __bf16* p = Bt + (n0 + (lane & 15)) * LDB + k0 + ((lane & 16) ? 8 : 0);
  union { unsigned long long u[4]; v16bf v; } U;
  U.u[0] = *(const unsigned long long*)(p);
  U.u[1] = *(const unsigned long long*)(p + 4);
  U.u[2] = *(const unsigned long long*)(p + 16);
  U.u[3] = *(const unsigned long long*)(p + 20);
  return U.v;
}
// C/D f32 16x16: VGPR r, lanes 0-15: M=r, N=lane; lanes 16-31: M=r+8, N=lane-16
__device__ __forceinline__ void store_c(float* C, int ld, int m0, int n0, v8f acc, int lane){
  int col = n0 + (lane & 15);
  int mb  = m0 + ((lane & 16) ? 8 : 0);
#pragma unroll
  for (int r = 0; r < 8; ++r) C[(mb + r) * ld + col] = acc[r];
}

// Complex 64x64 GEMM, K = 32*ksteps. A row-major bf16 (ld=LDB); B given transposed (Bt, ld=LDB).
// C f32 (ld=64). 16 output tiles over 8 waves (2 tiles/wave).
//   Cre = Ar*Br - Ai*Bi ; Cim = Ar*Bi + Ai*Br  (4 WMMAs per k-step per tile)
__device__ __forceinline__ void cgemm64(const __bf16* Ar, const __bf16* Ai,
                                        const __bf16* Btr, const __bf16* Bti,
                                        float* Cr, float* Ci, int ksteps,
                                        int lane, int wave){
#pragma unroll
  for (int t = 0; t < 2; ++t){
    int tile = wave * 2 + t;
    int m0 = (tile >> 2) << 4, n0 = (tile & 3) << 4;
    v8f aP = {}, aM = {}, aI = {};
    for (int ks = 0; ks < ksteps; ++ks){
      int k0 = ks << 5;
      v16bf far_ = frag_a (Ar,  m0, k0, lane);
      v16bf fai_ = frag_a (Ai,  m0, k0, lane);
      v16bf fbr_ = frag_bt(Btr, n0, k0, lane);
      v16bf fbi_ = frag_bt(Bti, n0, k0, lane);
      aP = __builtin_amdgcn_wmma_f32_16x16x32_bf16(false, far_, false, fbr_, (short)0, aP, false, false);
      aM = __builtin_amdgcn_wmma_f32_16x16x32_bf16(false, fai_, false, fbi_, (short)0, aM, false, false);
      aI = __builtin_amdgcn_wmma_f32_16x16x32_bf16(false, far_, false, fbi_, (short)0, aI, false, false);
      aI = __builtin_amdgcn_wmma_f32_16x16x32_bf16(false, fai_, false, fbr_, (short)0, aI, false, false);
    }
    store_c(Cr, 64, m0, n0, aP - aM, lane);
    store_c(Ci, 64, m0, n0, aI, lane);
  }
}

__global__ __launch_bounds__(256) void iaidnn_kernel(
    const float* __restrict__ Hg, const float* __restrict__ Vg,
    const float* __restrict__ aw, const float* __restrict__ pu,
    const float* __restrict__ ou, const float* __restrict__ pw,
    const float* __restrict__ pv, const float* __restrict__ ov,
    float* __restrict__ out)
{
  const int b    = blockIdx.x;
  const int tid  = threadIdx.x;
  const int lane = tid & 31, wave = tid >> 5;

  // big GEMM operand/result buffers (bf16 rows padded to LDB)
  __shared__ __attribute__((aligned(16))) __bf16 sHr[64*LDB], sHi[64*LDB]; // H_all row-major (A operand)
  __shared__ __attribute__((aligned(16))) __bf16 sGr[64*LDB], sGi[64*LDB]; // H_all transposed (B^T operand)
  __shared__ __attribute__((aligned(16))) __bf16 sAr[64*LDB], sAi[64*LDB]; // A: V_all -> P_all -> Av
  __shared__ __attribute__((aligned(16))) __bf16 sBr[64*LDB], sBi[64*LDB]; // B^T: conj(V)->Q^T->Yv^T->Mv
  __shared__ __attribute__((aligned(16))) __bf16 sVtr[16*LDB], sVti[16*LDB]; // Bv^T zero-padded 16x64
  __shared__ float  sCr[64][64], sCi[64][64];   // f32 GEMM out: Q -> HQ -> Av -> AvYv -> Vt
  // small per-k state ([0]=re, [1]=im)
  __shared__ float sAu[2][16][4][4];
  __shared__ float sHV[2][16][4][2];
  __shared__ float sM4[2][16][4][4];
  __shared__ float sU [2][16][4][2];
  __shared__ float sE [2][16][2][2];
  __shared__ float sW [2][16][2][2];
  __shared__ float sUW[2][16][4][2];
  __shared__ float sT [2][16][4][4];
  __shared__ float sInvdU[2][16][4];
  __shared__ float sItrU[2][16];
  __shared__ float sDg[2][64];
  __shared__ float sItrV[2];
  __shared__ float sVt[2][16][64][2];
  __shared__ float sAk[16];
  __shared__ float sMu;
  __shared__ float sRed[16];
  __shared__ float sPs[256];

  // ---------------- Phase 0: stage H_all (both orientations), V_all, conj(V) ----------------
  if (tid < 16) sAk[tid] = aw[tid];
  for (int idx = tid; idx < 4096; idx += 256){
    int m = idx >> 6, t = idx & 63;
    int k = m >> 2, r = m & 3;
    long gi = ((((long)b*16 + k)*4 + r)*64 + t)*2;
    float hr = Hg[gi], hi = Hg[gi+1];
    sHr[m*LDB + t] = (__bf16)hr;  sHi[m*LDB + t] = (__bf16)hi;
    sGr[t*LDB + m] = (__bf16)hr;  sGi[t*LDB + m] = (__bf16)hi;
  }
  for (int idx = tid; idx < 2048; idx += 256){
    int t = idx >> 5, c = idx & 31;          // c = k*2+s
    int k = c >> 1, s = c & 1;
    long gi = ((((long)b*16 + k)*64 + t)*2 + s)*2;
    float vr = Vg[gi], vi = Vg[gi+1];
    sAr[t*LDB + c] = (__bf16)vr;  sAi[t*LDB + c] = (__bf16)vi;    // V_all [64 x 32]
    sBr[t*LDB + c] = (__bf16)vr;  sBi[t*LDB + c] = (__bf16)(-vi); // Bt = conj(V) [n=64 x k=32]
  }
  __syncthreads();

  // ---------------- Phase 1: Q = V_all @ V_all^H  (64x64, K=32) ----------------
  cgemm64(sAr, sAi, sBr, sBi, &sCr[0][0], &sCi[0][0], 1, lane, wave);
  __syncthreads();
  for (int idx = tid; idx < 4096; idx += 256){   // Q -> bf16 B^T operand (transpose at write)
    int t = idx >> 6, n = idx & 63;              // read sCr rows contiguously
    sBr[n*LDB + t] = (__bf16)sCr[t][n];
    sBi[n*LDB + t] = (__bf16)sCi[t][n];
  }
  __syncthreads();

  // ---------------- Phase 2: HQ = H_all @ Q  (64x64, K=64) ----------------
  cgemm64(sHr, sHi, sBr, sBi, &sCr[0][0], &sCi[0][0], 2, lane, wave);
  __syncthreads();

  // ---------------- Phase 3: small per-k algebra (VALU f32) ----------------
  { // Au[k][r][j] = sum_t HQ[4k+r][t] conj(H[4k+j][t]) + noise*I
    int k = tid >> 4, r = (tid >> 2) & 3, j = tid & 3;
    cpx acc = { (r==j) ? NOISE_POW : 0.0f, 0.0f };
    for (int t2 = 0; t2 < 64; ++t2){
      cpx hq = { sCr[4*k + r][t2], sCi[4*k + r][t2] };
      cpx h  = { (float)sHr[(4*k + j)*LDB + t2], (float)sHi[(4*k + j)*LDB + t2] };
      acc.r += hq.r*h.r + hq.i*h.i;
      acc.i += hq.i*h.r - hq.r*h.i;
    }
    sAu[0][k][r][j] = acc.r; sAu[1][k][r][j] = acc.i;
  }
  if (tid < 128){ // HV[k][r][s] = sum_t H[4k+r][t] V[t][2k+s]
    int k = tid >> 3, r = (tid >> 1) & 3, s = tid & 1;
    cpx acc = {0,0};
    for (int t2 = 0; t2 < 64; ++t2){
      cpx h = { (float)sHr[(4*k + r)*LDB + t2], (float)sHi[(4*k + r)*LDB + t2] };
      cpx v = { (float)sAr[t2*LDB + 2*k + s], (float)sAi[t2*LDB + 2*k + s] };
      acc = cadd(acc, cmul(h, v));
    }
    sHV[0][k][r][s] = acc.r; sHV[1][k][r][s] = acc.i;
  }
  __syncthreads();

  { // Au @ Yu -> sM4
    int k = tid >> 4, r = (tid >> 2) & 3, j = tid & 3;
    cpx acc = {0,0};
    for (int p = 0; p < 4; ++p){
      cpx a = { sAu[0][k][r][p], sAu[1][k][r][p] };
      long yi = ((((long)1*16 + k)*4 + p)*4 + j)*2;
      cpx y = { pu[yi], pu[yi+1] };
      acc = cadd(acc, cmul(a, y));
    }
    sM4[0][k][r][j] = acc.r; sM4[1][k][r][j] = acc.i;
  }
  if (tid < 64){ // invd of diag(Au)
    int k = tid >> 2, r = tid & 3;
    cpx id = crecip_eps({ sAu[0][k][r][r], sAu[1][k][r][r] });
    sInvdU[0][k][r] = id.r; sInvdU[1][k][r] = id.i;
  }
  if (tid < 16){ // 1/(tr(Au)+eps)
    cpx tr = {0,0};
    for (int r = 0; r < 4; ++r){ tr.r += sAu[0][tid][r][r]; tr.i += sAu[1][tid][r][r]; }
    cpx it = crecip_eps(tr);
    sItrU[0][tid] = it.r; sItrU[1][tid] = it.i;
  }
  __syncthreads();

  { // Mu = invd*Xu + (Au@Yu)/tr + Zu  (in place in sM4)
    int k = tid >> 4, r = (tid >> 2) & 3, j = tid & 3;
    cpx ayu = { sM4[0][k][r][j], sM4[1][k][r][j] };
    long xi = ((((long)0*16 + k)*4 + r)*4 + j)*2;
    long zi = ((((long)2*16 + k)*4 + r)*4 + j)*2;
    cpx x = { pu[xi], pu[xi+1] }, z = { pu[zi], pu[zi+1] };
    cpx id = { sInvdU[0][k][r], sInvdU[1][k][r] };
    cpx it = { sItrU[0][k], sItrU[1][k] };
    cpx m = cadd(cadd(cmul(id, x), cmul(ayu, it)), z);
    sM4[0][k][r][j] = m.r; sM4[1][k][r][j] = m.i;
  }
  __syncthreads();

  if (tid < 128){ // U = Mu @ HV + Ou
    int k = tid >> 3, r = (tid >> 1) & 3, s = tid & 1;
    long oi = (((long)k*4 + r)*2 + s)*2;
    cpx acc = { ou[oi], ou[oi+1] };
    for (int j = 0; j < 4; ++j){
      cpx m  = { sM4[0][k][r][j], sM4[1][k][r][j] };
      cpx hv = { sHV[0][k][j][s], sHV[1][k][j][s] };
      acc = cadd(acc, cmul(m, hv));
    }
    sU[0][k][r][s] = acc.r; sU[1][k][r][s] = acc.i;
  }
  __syncthreads();

  if (tid < 64){ // E = I - U^H HV
    int k = tid >> 2, s1 = (tid >> 1) & 1, s2 = tid & 1;
    cpx acc = { (s1==s2) ? 1.0f : 0.0f, 0.0f };
    for (int r = 0; r < 4; ++r){
      cpx p = cmulc({ sU[0][k][r][s1], sU[1][k][r][s1] },
                    { sHV[0][k][r][s2], sHV[1][k][r][s2] });
      acc.r -= p.r; acc.i -= p.i;
    }
    sE[0][k][s1][s2] = acc.r; sE[1][k][s1][s2] = acc.i;
  }
  __syncthreads();

  if (tid < 64){ // W = approx_inv(E, Xw, Yw, Zw)
    int k = tid >> 2, i2 = (tid >> 1) & 1, j = tid & 1;
    cpx id = crecip_eps({ sE[0][k][i2][i2], sE[1][k][i2][i2] });
    cpx it = crecip_eps({ sE[0][k][0][0] + sE[0][k][1][1],
                          sE[1][k][0][0] + sE[1][k][1][1] });
    cpx eyw = {0,0};
    for (int p = 0; p < 2; ++p){
      long yi = ((((long)1*16 + k)*2 + p)*2 + j)*2;
      eyw = cadd(eyw, cmul({ sE[0][k][i2][p], sE[1][k][i2][p] }, { pw[yi], pw[yi+1] }));
    }
    long xi = ((((long)0*16 + k)*2 + i2)*2 + j)*2;
    long zi = ((((long)2*16 + k)*2 + i2)*2 + j)*2;
    cpx w = cadd(cadd(cmul(id, { pw[xi], pw[xi+1] }), cmul(eyw, it)), { pw[zi], pw[zi+1] });
    sW[0][k][i2][j] = w.r; sW[1][k][i2][j] = w.i;
  }
  __syncthreads();

  if (tid < 128){ // UW = U @ W
    int k = tid >> 3, r = (tid >> 1) & 3, s = tid & 1;
    cpx acc = {0,0};
    for (int j = 0; j < 2; ++j)
      acc = cadd(acc, cmul({ sU[0][k][r][j], sU[1][k][r][j] },
                           { sW[0][k][j][s], sW[1][k][j][s] }));
    sUW[0][k][r][s] = acc.r; sUW[1][k][r][s] = acc.i;
  }
  __syncthreads();

  { // T = a_k * UW U^H
    int k = tid >> 4, r = (tid >> 2) & 3, j = tid & 3;
    cpx acc = {0,0};
    for (int s = 0; s < 2; ++s){
      cpx uw = { sUW[0][k][r][s], sUW[1][k][r][s] };
      cpx u  = { sU[0][k][j][s],  sU[1][k][j][s]  };
      acc.r += uw.r*u.r + uw.i*u.i;   // uw * conj(u)
      acc.i += uw.i*u.r - uw.r*u.i;
    }
    sT[0][k][r][j] = sAk[k]*acc.r; sT[1][k][r][j] = sAk[k]*acc.i;
  }
  if (tid < 16){ // mu partials: a_k * Re(sum W[s][m] G[m][s]),  G = U^H U
    int k = tid; float part = 0.0f;
    for (int s = 0; s < 2; ++s)
      for (int m = 0; m < 2; ++m){
        cpx g = {0,0};
        for (int r = 0; r < 4; ++r)
          g = cadd(g, cmulc({ sU[0][k][r][m], sU[1][k][r][m] },
                            { sU[0][k][r][s], sU[1][k][r][s] }));
        cpx w = { sW[0][k][s][m], sW[1][k][s][m] };
        part += w.r*g.r - w.i*g.i;
      }
    sRed[k] = sAk[k] * part;
  }
  __syncthreads();
  if (tid == 0){
    float s = 0; for (int k = 0; k < 16; ++k) s += sRed[k];
    sMu = NOISE_POW * s / BSS_POW;
  }
  __syncthreads();

  // P_all[t][4k+j] = sum_r conj(H[4k+r][t]) T[k][r][j]  -> A operand (overwrites V_all)
  for (int idx = tid; idx < 4096; idx += 256){
    int t = idx >> 6, c = idx & 63;
    int k = c >> 2, j = c & 3;
    cpx acc = {0,0};
    for (int r = 0; r < 4; ++r)
      acc = cadd(acc, cmulc({ (float)sHr[(4*k + r)*LDB + t], (float)sHi[(4*k + r)*LDB + t] },
                            { sT[0][k][r][j], sT[1][k][r][j] }));
    sAr[t*LDB + c] = (__bf16)acc.r; sAi[t*LDB + c] = (__bf16)acc.i;
  }
  __syncthreads();

  // ---------------- Phase 4: Av = P_all @ H_all + mu*I  (B^T = H_all transposed = sG) ----------------
  cgemm64(sAr, sAi, sGr, sGi, &sCr[0][0], &sCi[0][0], 2, lane, wave);
  __syncthreads();
  if (tid < 64) sCr[tid][tid] += sMu;
  __syncthreads();
  if (tid < 64){
    cpx id = crecip_eps({ sCr[tid][tid], sCi[tid][tid] });
    sDg[0][tid] = id.r; sDg[1][tid] = id.i;
  }
  if (tid == 0){
    cpx tr = {0,0};
    for (int t2 = 0; t2 < 64; ++t2){ tr.r += sCr[t2][t2]; tr.i += sCi[t2][t2]; }
    cpx it = crecip_eps(tr);
    sItrV[0] = it.r; sItrV[1] = it.i;
  }
  for (int idx = tid; idx < 4096; idx += 256){   // Av -> bf16 A operand (persists over k loop)
    int t = idx >> 6, m = idx & 63;
    sAr[t*LDB + m] = (__bf16)sCr[t][m];
    sAi[t*LDB + m] = (__bf16)sCi[t][m];
  }
  for (int idx = tid; idx < 16*LDB; idx += 256){ // zero Bv^T pad (rows 2..15 stay zero)
    sVtr[idx] = (__bf16)0.0f;
    sVti[idx] = (__bf16)0.0f;
  }
  __syncthreads();

  // ---------------- Phase 5: per-k  Mv = approx_inv(Av)_k,  Vt = Mv@Bv + Ov ----------------
  float myp = 0.0f;
  for (int k = 0; k < 16; ++k){
    for (int idx = tid; idx < 4096; idx += 256){  // stage Yv_k (L2 resident) transposed as B^T
      long gi = (((long)(16 + k))*4096 + idx)*2;  // params_v[1][k], coalesced global read
      int p = idx >> 6, j = idx & 63;
      float yr = pv[gi], yi2 = pv[gi+1];
      sBr[j*LDB + p] = (__bf16)yr;                // Bt[n=j][k=p] = Yv[p][j]
      sBi[j*LDB + p] = (__bf16)yi2;
    }
    __syncthreads();
    cgemm64(sAr, sAi, sBr, sBi, &sCr[0][0], &sCi[0][0], 2, lane, wave);
    __syncthreads();
    cpx it = { sItrV[0], sItrV[1] };
    for (int idx = tid; idx < 4096; idx += 256){  // Mv = invd*Xv + AvYv/tr + Zv -> bf16 A operand
      int t = idx >> 6, m = idx & 63;             // (overwrites Yv B^T buffer, dead after GEMM)
      cpx ay = { sCr[t][m], sCi[t][m] };
      long xi = (((long)(0  + k))*4096 + idx)*2;
      long zi = (((long)(32 + k))*4096 + idx)*2;
      cpx x = { pv[xi], pv[xi+1] }, z = { pv[zi], pv[zi+1] };
      cpx id = { sDg[0][t], sDg[1][t] };
      cpx mv = cadd(cadd(cmul(id, x), cmul(ay, it)), z);
      sBr[t*LDB + m] = (__bf16)mv.r;              // Mv row-major [t][m]
      sBi[t*LDB + m] = (__bf16)mv.i;
    }
    if (tid < 128){ // Bv^T[s][t] = a_k * sum_r conj(H[4k+r][t]) UW[k][r][s]  (bf16, rows 0..1)
      int t = tid >> 1, s = tid & 1;
      cpx acc = {0,0};
      for (int r = 0; r < 4; ++r)
        acc = cadd(acc, cmulc({ (float)sHr[(4*k + r)*LDB + t], (float)sHi[(4*k + r)*LDB + t] },
                              { sUW[0][k][r][s], sUW[1][k][r][s] }));
      sVtr[s*LDB + t] = (__bf16)(sAk[k]*acc.r);
      sVti[s*LDB + t] = (__bf16)(sAk[k]*acc.i);
    }
    __syncthreads();
    // Vt_raw = Mv @ Bv on WMMA: 4 row tiles (M=64), one padded N=16 tile, K=64. Waves 0-3 only
    // (wave-uniform branch keeps EXEC all-ones inside participating waves).
    if (wave < 4){
      int m0 = wave << 4;
      v8f aP = {}, aM = {}, aI = {};
#pragma unroll
      for (int ks = 0; ks < 2; ++ks){
        int k0 = ks << 5;
        v16bf far_ = frag_a (sBr,  m0, k0, lane);   // Mv
        v16bf fai_ = frag_a (sBi,  m0, k0, lane);
        v16bf fbr_ = frag_bt(sVtr, 0,  k0, lane);   // Bv^T (zero-padded 16x64)
        v16bf fbi_ = frag_bt(sVti, 0,  k0, lane);
        aP = __builtin_amdgcn_wmma_f32_16x16x32_bf16(false, far_, false, fbr_, (short)0, aP, false, false);
        aM = __builtin_amdgcn_wmma_f32_16x16x32_bf16(false, fai_, false, fbi_, (short)0, aM, false, false);
        aI = __builtin_amdgcn_wmma_f32_16x16x32_bf16(false, far_, false, fbi_, (short)0, aI, false, false);
        aI = __builtin_amdgcn_wmma_f32_16x16x32_bf16(false, fai_, false, fbr_, (short)0, aI, false, false);
      }
      store_c(&sCr[0][0], 64, m0, 0, aP - aM, lane);
      store_c(&sCi[0][0], 64, m0, 0, aI, lane);
    }
    __syncthreads();
    if (tid < 128){ // Vt = Vt_raw + Ov ; accumulate power
      int t = tid >> 1, s = tid & 1;
      long oi = ((((long)k)*64 + t)*2 + s)*2;
      float vr = sCr[t][s] + ov[oi];
      float vi2 = sCi[t][s] + ov[oi+1];
      sVt[0][k][t][s] = vr; sVt[1][k][t][s] = vi2;
      myp += vr*vr + vi2*vi2;
    }
    __syncthreads();
  }

  // ---------------- Phase 6: power normalization + store ----------------
  sPs[tid] = myp;
  __syncthreads();
  for (int off = 128; off > 0; off >>= 1){
    if (tid < off) sPs[tid] += sPs[tid + off];
    __syncthreads();
  }
  float scale = sqrtf(BSS_POW / (sPs[0] + EPSV));
  for (int idx = tid; idx < 2048; idx += 256){
    int k = idx >> 7, t = (idx >> 1) & 63, s = idx & 1;
    long oi = ((((long)b*16 + k)*64 + t)*2 + s)*2;
    out[oi]   = scale * sVt[0][k][t][s];
    out[oi+1] = scale * sVt[1][k][t][s];
  }
}

extern "C" void kernel_launch(void* const* d_in, const int* in_sizes, int n_in,
                              void* d_out, int out_size, void* d_ws, size_t ws_size,
                              hipStream_t stream){
  (void)in_sizes; (void)n_in; (void)out_size; (void)d_ws; (void)ws_size;
  const float* Hg = (const float*)d_in[0];
  const float* Vg = (const float*)d_in[1];
  const float* aw = (const float*)d_in[2];
  const float* pu = (const float*)d_in[3];
  const float* ou = (const float*)d_in[4];
  const float* pw = (const float*)d_in[5];
  const float* pv = (const float*)d_in[6];
  const float* ov = (const float*)d_in[7];
  iaidnn_kernel<<<NBATCH, 256, 0, stream>>>(Hg, Vg, aw, pu, ou, pw, pv, ov, (float*)d_out);
}
